// CustomPatchEmbedding_20933670600872
// MI455X (gfx1250) — matrix operand order repaired and checked by reading, not compiled, not run on hardware
//
#include <hip/hip_runtime.h>

// ---------------------------------------------------------------------------
// Patch-embedding = gather + GEMM:
//   out[b,n,e] = sum_k patch[b,n,k] * W[e,k] + bias[e]
//   M = 32*576 = 18432, K = 3*16*16 = 768, E = 768
// MI455X roofline: 115 MB traffic -> ~5us floor @23.3TB/s. FP32 WMMA would be
// ~10x compute-bound; f16-in/f32-acc WMMA (16x16x32) needs ~1.33M WMMAs,
// matching the memory roofline. So: convert W to f16 once, gather patches to
// LDS as f16, and run v_wmma_f32_16x16x32_f16 tiles.
// ---------------------------------------------------------------------------

typedef __attribute__((ext_vector_type(16))) _Float16 v16h;
typedef __attribute__((ext_vector_type(8)))  _Float16 v8h;
typedef __attribute__((ext_vector_type(8)))  float    v8f;

#define PP   16
#define CC   3
#define HH   384
#define WW   384
#define BB   32
#define NN   576
#define EE   768
#define KK   768            // CC*PP*PP
#define MM   (BB * NN)      // 18432

#define BM   128            // block tile rows (patches)
#define BN   128            // block tile cols (embed dim)
#define KST  32             // K step (one wmma_f16 K)
#define LDSK 48             // padded LDS row stride in f16 (96B, 16B-aligned)

// --- fp32 -> f16 weight conversion (E*K = 589824 = 2304 * 256 exactly) -----
__global__ void wcvt_kernel(const float* __restrict__ w,
                            _Float16* __restrict__ w16) {
  int i = blockIdx.x * 256 + threadIdx.x;
  w16[i] = (_Float16)w[i];
}

// --- main fused gather + GEMM kernel ---------------------------------------
__global__ __launch_bounds__(256)
void patch_embed_wmma_kernel(const float* __restrict__ x,
                             const int*   __restrict__ centers,
                             const _Float16* __restrict__ w16,
                             const float* __restrict__ bias,
                             float* __restrict__ out) {
  __shared__ __align__(16) _Float16 As[BM * LDSK];

  const int tid    = threadIdx.x;
  const int lane   = tid & 31;            // wave32
  const int wave   = tid >> 5;            // 8 waves / block
  const int rowBlk = blockIdx.x * BM;     // patch-row base
  const int colBlk = blockIdx.y * BN;     // embed-col base

  // ---- staging role: each thread owns one 16-float image-row segment ----
  const int sRow  = tid >> 1;             // 0..127 local patch row
  const int sHalf = tid & 1;              // k-local [0,16) vs [16,32)
  const int m     = rowBlk + sRow;        // global patch index
  const int bImg  = m / NN;
  const int tok   = m - bImg * NN;
  const int chC   = centers[(bImg * NN + tok) * 2 + 0] - PP / 2;  // top row
  const int cwC   = centers[(bImg * NN + tok) * 2 + 1] - PP / 2;  // left col
  const float* srcBase =
      x + (size_t)bImg * (CC * HH * WW) + (size_t)chC * WW + cwC;

  // ---- wave sub-tile: 32 rows x 64 cols (2 x 4 wmma tiles) ----
  const int waveM = (wave >> 1) * 32;     // 0,32,64,96
  const int waveN = (wave & 1) * 64;      // 0,64
  const int fRow  = lane & 15;
  const int ksA   = (lane & 16);          // A frag K sub-offset: 0 or 16
  const int kbB   = (lane & 16);          // B frag K sub-offset: 0 or 16

  v8f acc[2][4];
#pragma unroll
  for (int mi = 0; mi < 2; ++mi)
#pragma unroll
    for (int ni = 0; ni < 4; ++ni)
      acc[mi][ni] = (v8f){0.f, 0.f, 0.f, 0.f, 0.f, 0.f, 0.f, 0.f};

  for (int kk = 0; kk < KK; kk += KST) {
    __syncthreads();  // previous iteration's LDS reads complete (WAR)

    // ---- gather+convert one 128x32 A chunk into LDS ----
    {
      const int c  = kk >> 8;                       // channel
      const int ph = ((kk & 255) >> 4) + sHalf;     // patch row in channel
      const float* src = srcBase + c * (HH * WW) + ph * WW;
      v8h h0, h1;
#pragma unroll
      for (int j = 0; j < 8; ++j) h0[j] = (_Float16)src[j];
#pragma unroll
      for (int j = 0; j < 8; ++j) h1[j] = (_Float16)src[8 + j];
      _Float16* dst = &As[sRow * LDSK + sHalf * 16];
      *(v8h*)(dst)     = h0;
      *(v8h*)(dst + 8) = h1;
    }
    __syncthreads();

    // ---- A fragments from LDS (contiguous k = 16h + s order, same as B) ----
    v16h a[2];
#pragma unroll
    for (int mi = 0; mi < 2; ++mi) {
      const _Float16* ap = &As[(waveM + mi * 16 + fRow) * LDSK];
      v8h lo = *(const v8h*)(ap + ksA);        // K = ksA + 0..7
      v8h hi = *(const v8h*)(ap + ksA + 8);    // K = ksA + 8..15
      a[mi] = __builtin_shufflevector(lo, hi, 0, 1, 2, 3, 4, 5, 6, 7, 8, 9,
                                      10, 11, 12, 13, 14, 15);
    }

    // ---- B fragments straight from L2-resident f16 weights ----
#pragma unroll
    for (int ni = 0; ni < 4; ++ni) {
      const int e = colBlk + waveN + ni * 16 + fRow;   // output column
      const _Float16* bp = w16 + (size_t)e * KK + kk + kbB;
      v8h lo = *(const v8h*)(bp);
      v8h hi = *(const v8h*)(bp + 8);
      v16h bf = __builtin_shufflevector(lo, hi, 0, 1, 2, 3, 4, 5, 6, 7, 8, 9,
                                        10, 11, 12, 13, 14, 15);
      acc[0][ni] = __builtin_amdgcn_wmma_f32_16x16x32_f16(
          false, a[0], false, bf, (short)0, acc[0][ni], false, false);
      acc[1][ni] = __builtin_amdgcn_wmma_f32_16x16x32_f16(
          false, a[1], false, bf, (short)0, acc[1][ni], false, false);
    }
  }

  // ---- epilogue: bias + store per f32 C/D layout ----
  const int rOff = (lane & 16) ? 8 : 0;
#pragma unroll
  for (int ni = 0; ni < 4; ++ni) {
    const int col = colBlk + waveN + ni * 16 + fRow;
    const float bv = bias[col];
#pragma unroll
    for (int mi = 0; mi < 2; ++mi) {
      const int rbase = rowBlk + waveM + mi * 16 + rOff;
#pragma unroll
      for (int i = 0; i < 8; ++i)
        out[(size_t)(rbase + i) * EE + col] = acc[mi][ni][i] + bv;
    }
  }
}

extern "C" void kernel_launch(void* const* d_in, const int* in_sizes, int n_in,
                              void* d_out, int out_size, void* d_ws,
                              size_t ws_size, hipStream_t stream) {
  const float* x       = (const float*)d_in[0];
  const int*   centers = (const int*)d_in[1];
  const float* proj_w  = (const float*)d_in[2];
  const float* proj_b  = (const float*)d_in[3];
  float*       out     = (float*)d_out;

  _Float16* w16 = (_Float16*)d_ws;  // EE*KK*2 = ~1.13 MB scratch

  // 1) convert weights to f16 (deterministic, every launch)
  wcvt_kernel<<<(EE * KK) / 256, 256, 0, stream>>>(proj_w, w16);

  // 2) fused gather + WMMA GEMM: grid 144 x 6 tiles of 128x128
  dim3 grid(MM / BM, EE / BN);
  patch_embed_wmma_kernel<<<grid, 256, 0, stream>>>(x, centers, w16, proj_b,
                                                    out);
}
